// SelfAttention_v2_37151467110590
// MI455X (gfx1250) — matrix operand order repaired and checked by reading, not compiled, not run on hardware
//
#include <hip/hip_runtime.h>
#include <hip/hip_bf16.h>
#include <stdint.h>

typedef __bf16 bf16;
typedef __attribute__((ext_vector_type(16))) __bf16 v16bf;
typedef __attribute__((ext_vector_type(8)))  float  v8f;

#define S_LEN 8192
#define D_DIM 1024
constexpr float SM_SCALE = 0.03125f; // 1/sqrt(1024)

// ---------------------------------------------------------------------------
// helpers
// ---------------------------------------------------------------------------
__device__ __forceinline__ v8f wmma_bf16(v16bf a, v16bf b, v8f c) {
  // (neg_a, A, neg_b, B, c_mod, C, reuse_a, reuse_b)
  return __builtin_amdgcn_wmma_f32_16x16x32_bf16(false, a, false, b,
                                                 (short)0, c, false, false);
}

// Build a 16-element bf16 fragment from two 16-byte chunks (A-fragment pattern:
// K = kh*8 + 0..7  and  K = 16 + kh*8 + 0..7 are each contiguous in memory).
__device__ __forceinline__ v16bf ld_frag2(const bf16* p0, const bf16* p1) {
  union { uint4 u[2]; v16bf v; } t;
  t.u[0] = *(const uint4*)p0;
  t.u[1] = *(const uint4*)p1;
  return t.v;
}

// ---------------------------------------------------------------------------
// fp32 -> bf16 conversion (vectorized: 4 floats in, 8 bytes out per thread)
// ---------------------------------------------------------------------------
__global__ void cvt_f32_to_bf16(const float* __restrict__ src,
                                bf16* __restrict__ dst, int n4) {
  int i = blockIdx.x * blockDim.x + threadIdx.x;
  if (i >= n4) return;
  float4 f = ((const float4*)src)[i];
  union { bf16 h[4]; uint64_t u; } t;
  t.h[0] = (bf16)f.x; t.h[1] = (bf16)f.y;
  t.h[2] = (bf16)f.z; t.h[3] = (bf16)f.w;
  ((uint64_t*)dst)[i] = t.u;
}

// ---------------------------------------------------------------------------
// QKV projection: Y = xb @ W^T  (A = x rows, B-columns = W rows -> direct loads)
// per wave: 32(M) x 64(N) tile, K-loop step 32, 8 WMMAs per step
// grid = (S/32, 2, 3): z selects Wq/Wk/Wv and the destination buffer
// ---------------------------------------------------------------------------
__global__ void __launch_bounds__(256)
qkv_gemm(const bf16* __restrict__ xb, const bf16* __restrict__ WbAll,
         bf16* __restrict__ Qb, bf16* __restrict__ Kb, bf16* __restrict__ Vb) {
  const int which = blockIdx.z;
  const bf16* W = WbAll + (size_t)which * (D_DIM * D_DIM);
  bf16* Y = (which == 0) ? Qb : (which == 1) ? Kb : Vb;

  const int lane = threadIdx.x & 31;
  const int w    = threadIdx.x >> 5;
  const int mlo  = lane & 15;
  const int kh   = lane >> 4;              // 0 or 1 (K-half select)
  const int mo   = (lane & 16) ? 8 : 0;    // C/D row offset for upper lanes
  const int m0   = blockIdx.x * 32;
  const int n0   = blockIdx.y * 512 + w * 64;

  v8f acc[2][4] = {};
  const bf16* arow0 = xb + (size_t)(m0 + mlo)      * D_DIM + kh * 8;
  const bf16* arow1 = xb + (size_t)(m0 + 16 + mlo) * D_DIM + kh * 8;

  for (int k0 = 0; k0 < D_DIM; k0 += 32) {
    v16bf a0 = ld_frag2(arow0 + k0, arow0 + k0 + 16);
    v16bf a1 = ld_frag2(arow1 + k0, arow1 + k0 + 16);
#pragma unroll
    for (int j = 0; j < 4; ++j) {
      const bf16* bp = W + (size_t)(n0 + j * 16 + mlo) * D_DIM + k0 + kh * 8;
      v16bf b = ld_frag2(bp, bp + 16);
      acc[0][j] = wmma_bf16(a0, b, acc[0][j]);
      acc[1][j] = wmma_bf16(a1, b, acc[1][j]);
    }
  }

#pragma unroll
  for (int mt = 0; mt < 2; ++mt)
#pragma unroll
    for (int j = 0; j < 4; ++j) {
      const size_t row = (size_t)(m0 + mt * 16 + mo);
#pragma unroll
      for (int r = 0; r < 8; ++r)
        Y[(row + r) * D_DIM + n0 + j * 16 + mlo] = (bf16)acc[mt][j][r];
    }
}

// ---------------------------------------------------------------------------
// V transpose: Vb[8192][1024] -> Vt[1024][8192]  (B-columns for P@V become rows)
// ---------------------------------------------------------------------------
__global__ void transpose_bf16(const bf16* __restrict__ src,
                               bf16* __restrict__ dst) {
  __shared__ bf16 tile[32][33];
  const int sb = blockIdx.x * 32;           // seq block
  const int db = blockIdx.y * 32;           // dim block
  const int tx = threadIdx.x & 31;
  const int ty = threadIdx.x >> 5;          // 0..7
#pragma unroll
  for (int i = 0; i < 32; i += 8)
    tile[ty + i][tx] = src[(size_t)(sb + ty + i) * D_DIM + db + tx];
  __syncthreads();
#pragma unroll
  for (int i = 0; i < 32; i += 8)
    dst[(size_t)(db + ty + i) * S_LEN + sb + tx] = tile[tx][ty + i];
}

// ---------------------------------------------------------------------------
// Fused flash attention: block = 16 query rows, 8 waves.
//  - Q panel (16x1024 bf16, 32KB) async-staged into LDS once
//  - per 128-key tile: wave w computes a 16x16 score tile (K rows stream
//    through L2), online softmax in LDS, then each wave does its 128-wide
//    output slice O += P @ V with 32 WMMAs
// ---------------------------------------------------------------------------
__global__ void __launch_bounds__(256)
attention_fused(const bf16* __restrict__ Qb, const bf16* __restrict__ Kb,
                const bf16* __restrict__ Vt, float* __restrict__ out) {
  __shared__ __align__(16) bf16  Qs[16][D_DIM];   // 32 KB
  __shared__ __align__(16) float Ss[16][128];     // 8 KB
  __shared__ __align__(16) bf16  Ps[16][128];     // 4 KB
  __shared__ float red[16][16];
  __shared__ float m_row[16], l_row[16], scl[16];

  const int tid  = threadIdx.x;
  const int lane = tid & 31;
  const int w    = tid >> 5;                 // wave 0..7
  const int q0   = blockIdx.x * 16;
  const int mlo  = lane & 15;
  const int kh   = lane >> 4;
  const int mo   = (lane & 16) ? 8 : 0;

  if (tid < 16) { m_row[tid] = -3.0e38f; l_row[tid] = 0.0f; }

  // -- async stage the Q panel: 32KB = 2048 x b128, 8 per thread ------------
  {
    const bf16* qsrc = Qb + (size_t)q0 * D_DIM;
    unsigned base = (unsigned)(size_t)&Qs[0][0];
#pragma unroll
    for (int i = 0; i < 8; ++i) {
      unsigned off = (unsigned)(tid + i * 256) * 16u;
      unsigned lds_addr = base + off;
      unsigned long long gaddr = (unsigned long long)qsrc + off;
      asm volatile("global_load_async_to_lds_b128 %0, %1, off"
                   :: "v"(lds_addr), "v"(gaddr) : "memory");
    }
    asm volatile("s_wait_asynccnt 0x0" ::: "memory");
  }
  __syncthreads();

  v8f oacc[8] = {};

  for (int kv0 = 0; kv0 < S_LEN; kv0 += 128) {
    // ---- scores: wave w owns keys [kv0 + 16w, +16) ------------------------
    {
      const int j0 = kv0 + w * 16;
      if (kv0 + 128 < S_LEN)   // prefetch next tile's K rows into L2
        __builtin_prefetch(Kb + (size_t)(j0 + 128 + mlo) * D_DIM, 0, 1);
      v8f sacc = {};
      const bf16* krow = Kb + (size_t)(j0 + mlo) * D_DIM + kh * 8;
      for (int d0 = 0; d0 < D_DIM; d0 += 32) {
        v16bf a = ld_frag2(&Qs[mlo][d0 + kh * 8], &Qs[mlo][d0 + 16 + kh * 8]);
        v16bf b = ld_frag2(krow + d0, krow + d0 + 16);
        sacc = wmma_bf16(a, b, sacc);
      }
#pragma unroll
      for (int r = 0; r < 8; ++r)
        Ss[r + mo][w * 16 + mlo] = sacc[r] * SM_SCALE;
    }
    __syncthreads();

    // ---- online softmax over the 16x128 tile ------------------------------
    {
      const int r = tid & 15, c = tid >> 4;   // 16 rows x 16 chunks of 8 cols
      float lm = -3.0e38f;
#pragma unroll
      for (int i = 0; i < 8; ++i) lm = fmaxf(lm, Ss[r][c * 8 + i]);
      red[r][c] = lm;
      __syncthreads();
      if (tid < 16) {
        float mx = m_row[tid];
#pragma unroll
        for (int i = 0; i < 16; ++i) mx = fmaxf(mx, red[tid][i]);
        float sc = __expf(m_row[tid] - mx);
        scl[tid] = sc;
        l_row[tid] *= sc;
        m_row[tid] = mx;
      }
      __syncthreads();
      float ls = 0.0f;
#pragma unroll
      for (int i = 0; i < 8; ++i) {
        float p = __expf(Ss[r][c * 8 + i] - m_row[r]);
        Ps[r][c * 8 + i] = (bf16)p;
        ls += p;
      }
      red[r][c] = ls;
      __syncthreads();
      if (tid < 16) {
        float s = 0.0f;
#pragma unroll
        for (int i = 0; i < 16; ++i) s += red[tid][i];
        l_row[tid] += s;
      }
      __syncthreads();
    }

    // ---- O = diag(scl)*O + P @ V; wave w owns d-cols [128w, 128w+128) -----
    {
      float s8[8];
#pragma unroll
      for (int r = 0; r < 8; ++r) s8[r] = scl[r + mo];
#pragma unroll
      for (int nt = 0; nt < 8; ++nt)
#pragma unroll
        for (int r = 0; r < 8; ++r)
          oacc[nt][r] *= s8[r];

#pragma unroll
      for (int c4 = 0; c4 < 4; ++c4) {
        v16bf a = ld_frag2(&Ps[mlo][c4 * 32 + kh * 8],
                           &Ps[mlo][c4 * 32 + 16 + kh * 8]);
#pragma unroll
        for (int nt = 0; nt < 8; ++nt) {
          const bf16* vp = Vt + (size_t)(w * 128 + nt * 16 + mlo) * S_LEN
                              + kv0 + c4 * 32 + kh * 8;
          v16bf b = ld_frag2(vp, vp + 16);
          oacc[nt] = wmma_bf16(a, b, oacc[nt]);
        }
      }
    }
    __syncthreads();
  }

  // ---- normalize by the softmax denominator, store fp32 -------------------
  float linv[8];
#pragma unroll
  for (int r = 0; r < 8; ++r) linv[r] = 1.0f / l_row[r + mo];
#pragma unroll
  for (int nt = 0; nt < 8; ++nt) {
    const int dcol = w * 128 + nt * 16 + mlo;
#pragma unroll
    for (int r = 0; r < 8; ++r)
      out[(size_t)(q0 + r + mo) * D_DIM + dcol] = oacc[nt][r] * linv[r];
  }
}

// ---------------------------------------------------------------------------
// launch
// ---------------------------------------------------------------------------
extern "C" void kernel_launch(void* const* d_in, const int* in_sizes, int n_in,
                              void* d_out, int out_size, void* d_ws,
                              size_t ws_size, hipStream_t stream) {
  const float* x  = (const float*)d_in[0];
  const float* Wq = (const float*)d_in[1];
  const float* Wk = (const float*)d_in[2];
  const float* Wv = (const float*)d_in[3];
  float* out = (float*)d_out;

  char* ws = (char*)d_ws;
  const size_t MB = (size_t)1 << 20;
  bf16* xb = (bf16*)(ws);                 // 16 MB  x  [8192][1024] bf16
  bf16* Qb = (bf16*)(ws + 16 * MB);       // 16 MB
  bf16* Kb = (bf16*)(ws + 32 * MB);       // 16 MB
  bf16* Vb = (bf16*)(ws + 48 * MB);       // 16 MB
  bf16* Vt = (bf16*)(ws + 64 * MB);       // 16 MB  V^T [1024][8192]
  bf16* Wb = (bf16*)(ws + 80 * MB);       //  6 MB  [3][1024][1024]

  const int NX = S_LEN * D_DIM;           // 8M elems
  const int NW = D_DIM * D_DIM;           // 1M elems

  cvt_f32_to_bf16<<<NX / 4 / 256, 256, 0, stream>>>(x,  xb, NX / 4);
  cvt_f32_to_bf16<<<NW / 4 / 256, 256, 0, stream>>>(Wq, Wb,          NW / 4);
  cvt_f32_to_bf16<<<NW / 4 / 256, 256, 0, stream>>>(Wk, Wb + NW,     NW / 4);
  cvt_f32_to_bf16<<<NW / 4 / 256, 256, 0, stream>>>(Wv, Wb + 2 * NW, NW / 4);

  qkv_gemm<<<dim3(S_LEN / 32, 2, 3), 256, 0, stream>>>(xb, Wb, Qb, Kb, Vb);
  transpose_bf16<<<dim3(S_LEN / 32, D_DIM / 32), 256, 0, stream>>>(Vb, Vt);
  attention_fused<<<S_LEN / 16, 256, 0, stream>>>(Qb, Kb, Vt, out);
}
